// TypedDualBankSharedMoEFFN_27384711479466
// MI455X (gfx1250) — matrix operand order repaired and checked by reading, not compiled, not run on hardware
//
#include <hip/hip_runtime.h>
#include <math.h>
#include <stdint.h>

// ---------------------------------------------------------------------------
// Problem constants (from reference)
// ---------------------------------------------------------------------------
#define BB   16
#define CC   32
#define SS   64
#define DD   512
#define TT   (CC * SS)      // 2048 tokens per sample
#define DFF  2048
#define EE   8

// GEMM tiling
#define MTILE  128          // workgroup rows (4 waves x 32)
#define NTILE  64           // workgroup cols (weight tile staged in LDS)
#define KCHUNK 128          // K staged per LDS buffer (64 x 128 bf16 = 16 KB)

typedef __bf16 bf16_t;
typedef __attribute__((ext_vector_type(4)))  __bf16 v4bf;
typedef __attribute__((ext_vector_type(8)))  __bf16 v8bf;
typedef __attribute__((ext_vector_type(16))) __bf16 v16bf;
typedef __attribute__((ext_vector_type(8)))  float  v8f;

// ---------------------------------------------------------------------------
// WMMA helpers (CDNA5: v_wmma_f32_16x16x32_bf16, wave32)
// ---------------------------------------------------------------------------
__device__ __forceinline__ v8f wmma_bf16(v16bf a, v16bf b, v8f c) {
    // 8 args: (neg_a, A, neg_b, B, c_mod, C, reuse_a, reuse_b)
    return __builtin_amdgcn_wmma_f32_16x16x32_bf16(
        false, a, false, b, (short)0, c, false, false);
}

// A-matrix 16x32 bf16 fragment (row-major memory, ld = K stride).
// ISA layout: lanes 0-15 -> M=lane,    K = {0..7, 16..23}
//             lanes 16-31 -> M=lane-16, K = {8..15, 24..31}
__device__ __forceinline__ v16bf load_a_frag(const bf16_t* A, int lda,
                                             int mBase, int kBase, int lane) {
    int row = mBase + (lane & 15);
    int khi = (lane >> 4) << 3;                 // 0 or 8
    const bf16_t* p = A + (size_t)row * lda + kBase + khi;
    v8bf lo = *reinterpret_cast<const v8bf*>(p);        // K khi..khi+7
    v8bf hi = *reinterpret_cast<const v8bf*>(p + 16);   // K khi+16..khi+23
    return __builtin_shufflevector(lo, hi, 0, 1, 2, 3, 4, 5, 6, 7,
                                           8, 9, 10, 11, 12, 13, 14, 15);
}

// B-matrix 32x16 bf16 fragment from an LDS tile holding W rows [N, K]
// (row n of W == column n of B). ISA layout: lanes 0-15 -> N=lane, K={0..7,16..23};
// lanes 16-31 -> N=lane-16, K={8..15,24..31}.
__device__ __forceinline__ v16bf load_b_frag(const bf16_t* W, int ldb,
                                             int nBase, int kBase, int lane) {
    int row  = nBase + (lane & 15);
    int koff = (lane >> 4) << 3;                // 0 or 8
    const bf16_t* p = W + (size_t)row * ldb + kBase + koff;
    v8bf lo = *reinterpret_cast<const v8bf*>(p);
    v8bf hi = *reinterpret_cast<const v8bf*>(p + 16);
    return __builtin_shufflevector(lo, hi, 0, 1, 2, 3, 4, 5, 6, 7,
                                           8, 9, 10, 11, 12, 13, 14, 15);
}

// tanh-approx gelu via hardware exp:
// 0.5*x*(1+tanh(y)) == x / (1 + exp(-2y)),  y = c0*(x + c1*x^3)
__device__ __forceinline__ float gelu_tanh(float x) {
    const float c0 = 0.7978845608028654f;   // sqrt(2/pi)
    const float c1 = 0.044715f;
    float y = c0 * (x + c1 * x * x * x);
    return x / (1.0f + __expf(-2.0f * y));
}

// ---------------------------------------------------------------------------
// CDNA5 async copy: global -> LDS, 16 bytes per lane per issue, ASYNCcnt.
// GVS addressing: SGPR64 base + per-lane VGPR byte offset.
// ---------------------------------------------------------------------------
__device__ __forceinline__ void async_g2l_b128(uint32_t lds_off,
                                               uint64_t gbase, uint32_t goff) {
    asm volatile("global_load_async_to_lds_b128 %0, %1, %2"
                 :: "v"(lds_off), "v"(goff), "s"(gbase)
                 : "memory");
}
__device__ __forceinline__ void wait_async0() {
    asm volatile("s_wait_asynccnt 0x0" ::: "memory");
}

// Stage a 64-row x 128-col bf16 weight tile (16 KB) into LDS.
// 1024 x 16B units; 128 threads x 8 issues. LDS layout row-major with
// row stride KCHUNK, which equals the global per-row chunk size, so flat
// unit indexing matches on both sides.
__device__ __forceinline__ void stage_w_tile(uint32_t ldsBase, const bf16_t* W,
                                             int ldb, int nBase, int kBase,
                                             int tid) {
    uint64_t gb = (uint64_t)(uintptr_t)W;
    #pragma unroll
    for (int i = 0; i < 8; ++i) {
        int u    = i * 128 + tid;
        int row  = u >> 4;              // 16 units per 128-elem row
        int col8 = (u & 15) << 3;       // element offset within chunk
        uint32_t goff = (uint32_t)(((uint32_t)(nBase + row) * (uint32_t)ldb +
                                    (uint32_t)(kBase + col8)) * 2u);
        uint32_t loff = ldsBase + ((uint32_t)u << 4);
        async_g2l_b128(loff, gb, goff);
    }
}

// ---------------------------------------------------------------------------
// Kernel 1: per-(b,d) means of baseline / attnres over C*S positions
// ---------------------------------------------------------------------------
__global__ void means_kernel(const float* __restrict__ baseline,
                             const float* __restrict__ attnres,
                             float* __restrict__ bm, float* __restrict__ am) {
    int tid = blockIdx.x * blockDim.x + threadIdx.x;   // 0 .. B*D-1
    if (tid >= BB * DD) return;
    int b = tid / DD, d = tid % DD;
    const float* pb = baseline + (size_t)b * TT * DD + d;
    const float* pa = attnres  + (size_t)b * TT * DD + d;
    float sb = 0.f, sa = 0.f;
    for (int t = 0; t < TT; ++t) {
        sb += pb[(size_t)t * DD];
        sa += pa[(size_t)t * DD];
    }
    bm[tid] = sb * (1.0f / TT);
    am[tid] = sa * (1.0f / TT);
}

// ---------------------------------------------------------------------------
// Kernel 2: routers. 256 threads: thread = (b, router, expert) dot of 1536.
// ---------------------------------------------------------------------------
__global__ void route_kernel(const float* __restrict__ bm,
                             const float* __restrict__ am,
                             const float* __restrict__ spa_rw,
                             const float* __restrict__ spa_rb,
                             const float* __restrict__ spe_rw,
                             const float* __restrict__ spe_rb,
                             int* __restrict__ spaIdx, float* __restrict__ spaGate,
                             int* __restrict__ speIdx, float* __restrict__ speGate) {
    __shared__ float lg[BB * 2 * EE];
    int t = threadIdx.x;               // 0..255
    int b = t >> 4;
    int r = (t >> 3) & 1;
    int e = t & 7;
    const float* rw = (r == 0 ? spa_rw : spe_rw) + (size_t)e * (3 * DD);
    const float* rb = (r == 0 ? spa_rb : spe_rb);
    const float* bmb = bm + (size_t)b * DD;
    const float* amb = am + (size_t)b * DD;
    float s = rb[e];
    for (int j = 0; j < DD; ++j) {
        float bv = bmb[j], av = amb[j];
        s += bv * rw[j] + av * rw[DD + j] + (av - bv) * rw[2 * DD + j];
    }
    lg[t] = s;
    __syncthreads();
    if (t < BB * 2) {
        int b2 = t >> 1, r2 = t & 1;
        const float* L = &lg[b2 * 16 + r2 * 8];
        float mx = L[0]; int ai = 0;
        for (int i = 1; i < EE; ++i) if (L[i] > mx) { mx = L[i]; ai = i; }
        float den = 0.f;
        for (int i = 0; i < EE; ++i) den += expf(L[i] - mx);
        float gate = 1.0f / den;       // softmax value at the argmax
        if (r2 == 0) { spaIdx[b2] = ai; spaGate[b2] = gate; }
        else         { speIdx[b2] = ai; speGate[b2] = gate; }
    }
}

// ---------------------------------------------------------------------------
// Kernel 3: f32 -> bf16 conversion, 4 elements per thread (b128 in, b64 out)
// ---------------------------------------------------------------------------
__global__ void cvt_kernel(const float4* __restrict__ src,
                           v4bf* __restrict__ dst, size_t n4) {
    size_t i = (size_t)blockIdx.x * blockDim.x + threadIdx.x;
    size_t stride = (size_t)gridDim.x * blockDim.x;
    for (; i < n4; i += stride) {
        float4 f = src[i];
        v4bf o;
        o.x = (bf16_t)f.x; o.y = (bf16_t)f.y;
        o.z = (bf16_t)f.z; o.w = (bf16_t)f.w;
        dst[i] = o;
    }
}

// ---------------------------------------------------------------------------
// Kernel 4: GEMM1 + bias + gelu -> H (bf16)
//   H[b] = gelu(X[b] (TxD) @ W1 (DFFxD)^T + b1)
// grid: x = T/MTILE, y = DFF/NTILE, z = b ; block = 128 (4 waves)
// Weight tile double-buffered in LDS via global_load_async_to_lds_b128.
// ---------------------------------------------------------------------------
__global__ __launch_bounds__(128)
void ffn1_kernel(const bf16_t* __restrict__ X, const bf16_t* __restrict__ W1,
                 const int* __restrict__ idxArr, const float* __restrict__ b1,
                 bf16_t* __restrict__ H) {
    __shared__ alignas(16) bf16_t sm[2][NTILE * KCHUNK];
    int b = blockIdx.z;
    int widx = idxArr ? idxArr[b] : 0;
    const bf16_t* Wp = W1 + (size_t)widx * DFF * DD;
    const float*  bp = b1 + (size_t)widx * DFF;
    const bf16_t* Xp = X + (size_t)b * TT * DD;
    bf16_t*       Hp = H + (size_t)b * TT * DFF;

    int tid  = threadIdx.x;
    int lane = tid & 31;
    int wave = tid >> 5;
    int mWave = blockIdx.x * MTILE + wave * 32;  // 32 unique rows per wave
    int nTile = blockIdx.y * NTILE;

    uint32_t sm0 = (uint32_t)(uintptr_t)&sm[0][0];
    uint32_t sm1 = (uint32_t)(uintptr_t)&sm[1][0];

    v8f acc[2][4] = {};
    const int nChunks = DD / KCHUNK;             // 4
    stage_w_tile(sm0, Wp, DD, nTile, 0, tid);
    for (int kc = 0; kc < nChunks; ++kc) {
        wait_async0();
        __syncthreads();
        if (kc + 1 < nChunks)
            stage_w_tile(((kc + 1) & 1) ? sm1 : sm0, Wp, DD, nTile,
                         (kc + 1) * KCHUNK, tid);
        const bf16_t* smb = &sm[kc & 1][0];
        int kG = kc * KCHUNK;
        #pragma unroll
        for (int ks = 0; ks < KCHUNK; ks += 32) {
            v16bf a0 = load_a_frag(Xp, DD, mWave,      kG + ks, lane);
            v16bf a1 = load_a_frag(Xp, DD, mWave + 16, kG + ks, lane);
            #pragma unroll
            for (int nj = 0; nj < 4; ++nj) {
                v16bf wf = load_b_frag(smb, KCHUNK, nj * 16, ks, lane);
                acc[0][nj] = wmma_bf16(a0, wf, acc[0][nj]);
                acc[1][nj] = wmma_bf16(a1, wf, acc[1][nj]);
            }
        }
    }
    // C/D layout: VGPR r -> lanes 0-15: M=r, N=lane ; lanes 16-31: M=r+8
    int rowAdd = (lane >> 4) << 3;
    int colOff = lane & 15;
    #pragma unroll
    for (int mi = 0; mi < 2; ++mi)
        #pragma unroll
        for (int nj = 0; nj < 4; ++nj) {
            int colBase = nTile + nj * 16 + colOff;
            float bias = bp[colBase];
            #pragma unroll
            for (int r = 0; r < 8; ++r) {
                int row = mWave + mi * 16 + r + rowAdd;
                float v = acc[mi][nj][r] + bias;
                Hp[(size_t)row * DFF + colBase] = (bf16_t)gelu_tanh(v);
            }
        }
}

// ---------------------------------------------------------------------------
// Kernel 5: GEMM2 + bias + gate, accumulate into out (f32)
//   out[b] (init? = : +=) gate[b] * (H[b] (TxDFF) @ W2 (DxDFF)^T + b2)
// grid: x = T/MTILE, y = D/NTILE, z = b ; block = 128
// ---------------------------------------------------------------------------
__global__ __launch_bounds__(128)
void ffn2_kernel(const bf16_t* __restrict__ H, const bf16_t* __restrict__ W2,
                 const int* __restrict__ idxArr, const float* __restrict__ b2,
                 const float* __restrict__ gateArr, float* __restrict__ out,
                 int initFlag) {
    __shared__ alignas(16) bf16_t sm[2][NTILE * KCHUNK];
    int b = blockIdx.z;
    int widx = idxArr ? idxArr[b] : 0;
    const bf16_t* Wp = W2 + (size_t)widx * DD * DFF;
    const float*  bp = b2 + (size_t)widx * DD;
    const bf16_t* Hp = H + (size_t)b * TT * DFF;
    float*        Op = out + (size_t)b * TT * DD;
    float g = gateArr ? gateArr[b] : 1.0f;

    int tid  = threadIdx.x;
    int lane = tid & 31;
    int wave = tid >> 5;
    int mWave = blockIdx.x * MTILE + wave * 32;
    int nTile = blockIdx.y * NTILE;

    uint32_t sm0 = (uint32_t)(uintptr_t)&sm[0][0];
    uint32_t sm1 = (uint32_t)(uintptr_t)&sm[1][0];

    v8f acc[2][4] = {};
    const int nChunks = DFF / KCHUNK;            // 16
    stage_w_tile(sm0, Wp, DFF, nTile, 0, tid);
    for (int kc = 0; kc < nChunks; ++kc) {
        wait_async0();
        __syncthreads();
        if (kc + 1 < nChunks)
            stage_w_tile(((kc + 1) & 1) ? sm1 : sm0, Wp, DFF, nTile,
                         (kc + 1) * KCHUNK, tid);
        const bf16_t* smb = &sm[kc & 1][0];
        int kG = kc * KCHUNK;
        #pragma unroll
        for (int ks = 0; ks < KCHUNK; ks += 32) {
            v16bf a0 = load_a_frag(Hp, DFF, mWave,      kG + ks, lane);
            v16bf a1 = load_a_frag(Hp, DFF, mWave + 16, kG + ks, lane);
            #pragma unroll
            for (int nj = 0; nj < 4; ++nj) {
                v16bf wf = load_b_frag(smb, KCHUNK, nj * 16, ks, lane);
                acc[0][nj] = wmma_bf16(a0, wf, acc[0][nj]);
                acc[1][nj] = wmma_bf16(a1, wf, acc[1][nj]);
            }
        }
    }
    int rowAdd = (lane >> 4) << 3;
    int colOff = lane & 15;
    #pragma unroll
    for (int mi = 0; mi < 2; ++mi)
        #pragma unroll
        for (int nj = 0; nj < 4; ++nj) {
            int colBase = nTile + nj * 16 + colOff;
            float bias = bp[colBase];
            #pragma unroll
            for (int r = 0; r < 8; ++r) {
                int row = mWave + mi * 16 + r + rowAdd;
                size_t o = (size_t)row * DD + colBase;
                float v = g * (acc[mi][nj][r] + bias);
                Op[o] = initFlag ? v : (Op[o] + v);
            }
        }
}

// ---------------------------------------------------------------------------
// Host-side launcher
// ---------------------------------------------------------------------------
static inline size_t align256(size_t x) { return (x + 255) & ~(size_t)255; }

extern "C" void kernel_launch(void* const* d_in, const int* in_sizes, int n_in,
                              void* d_out, int out_size, void* d_ws, size_t ws_size,
                              hipStream_t stream) {
    const float* x          = (const float*)d_in[0];
    const float* baseline   = (const float*)d_in[1];
    const float* attnres    = (const float*)d_in[2];
    const float* shared_w1  = (const float*)d_in[3];
    const float* shared_b1  = (const float*)d_in[4];
    const float* shared_w2  = (const float*)d_in[5];
    const float* shared_b2  = (const float*)d_in[6];
    const float* spa_rw     = (const float*)d_in[7];
    const float* spa_rb     = (const float*)d_in[8];
    const float* spe_rw     = (const float*)d_in[9];
    const float* spe_rb     = (const float*)d_in[10];
    const float* spa_w1     = (const float*)d_in[11];
    const float* spa_b1     = (const float*)d_in[12];
    const float* spa_w2     = (const float*)d_in[13];
    const float* spa_b2     = (const float*)d_in[14];
    const float* spe_w1     = (const float*)d_in[15];
    const float* spe_b1     = (const float*)d_in[16];
    const float* spe_w2     = (const float*)d_in[17];
    const float* spe_b2     = (const float*)d_in[18];
    float* out = (float*)d_out;

    // Workspace layout
    char* ws = (char*)d_ws;
    size_t off = 0;
    float* bm      = (float*)(ws + off); off = align256(off + (size_t)BB * DD * 4);
    float* am      = (float*)(ws + off); off = align256(off + (size_t)BB * DD * 4);
    int*   spaIdx  = (int*)  (ws + off); off = align256(off + BB * 4);
    float* spaGate = (float*)(ws + off); off = align256(off + BB * 4);
    int*   speIdx  = (int*)  (ws + off); off = align256(off + BB * 4);
    float* speGate = (float*)(ws + off); off = align256(off + BB * 4);
    bf16_t* Xb     = (bf16_t*)(ws + off); off = align256(off + (size_t)BB * TT * DD * 2);
    bf16_t* w1sB   = (bf16_t*)(ws + off); off = align256(off + (size_t)DFF * DD * 2);
    bf16_t* w2sB   = (bf16_t*)(ws + off); off = align256(off + (size_t)DD * DFF * 2);
    bf16_t* spaW1B = (bf16_t*)(ws + off); off = align256(off + (size_t)EE * DFF * DD * 2);
    bf16_t* spaW2B = (bf16_t*)(ws + off); off = align256(off + (size_t)EE * DD * DFF * 2);
    bf16_t* speW1B = (bf16_t*)(ws + off); off = align256(off + (size_t)EE * DFF * DD * 2);
    bf16_t* speW2B = (bf16_t*)(ws + off); off = align256(off + (size_t)EE * DD * DFF * 2);
    bf16_t* Hbuf   = (bf16_t*)(ws + off); off = align256(off + (size_t)BB * TT * DFF * 2);
    (void)ws_size; (void)n_in; (void)in_sizes; (void)out_size;

    // 1) routing features + routing
    means_kernel<<<(BB * DD + 255) / 256, 256, 0, stream>>>(baseline, attnres, bm, am);
    route_kernel<<<1, 256, 0, stream>>>(bm, am, spa_rw, spa_rb, spe_rw, spe_rb,
                                        spaIdx, spaGate, speIdx, speGate);

    // 2) f32 -> bf16 conversions (4 elements per thread)
    cvt_kernel<<<1024, 256, 0, stream>>>((const float4*)x, (v4bf*)Xb,
                                         (size_t)BB * TT * DD / 4);
    cvt_kernel<<<128, 256, 0, stream>>>((const float4*)shared_w1, (v4bf*)w1sB,
                                        (size_t)DFF * DD / 4);
    cvt_kernel<<<128, 256, 0, stream>>>((const float4*)shared_w2, (v4bf*)w2sB,
                                        (size_t)DD * DFF / 4);
    cvt_kernel<<<1024, 256, 0, stream>>>((const float4*)spa_w1, (v4bf*)spaW1B,
                                         (size_t)EE * DFF * DD / 4);
    cvt_kernel<<<1024, 256, 0, stream>>>((const float4*)spa_w2, (v4bf*)spaW2B,
                                         (size_t)EE * DD * DFF / 4);
    cvt_kernel<<<1024, 256, 0, stream>>>((const float4*)spe_w1, (v4bf*)speW1B,
                                         (size_t)EE * DFF * DD / 4);
    cvt_kernel<<<1024, 256, 0, stream>>>((const float4*)spe_w2, (v4bf*)speW2B,
                                         (size_t)EE * DD * DFF / 4);

    dim3 g1(TT / MTILE, DFF / NTILE, BB);   // 16 x 32 x 16
    dim3 g2(TT / MTILE, DD / NTILE, BB);    // 16 x 8 x 16
    dim3 blk(128);

    // 3) shared branch (gate = 1, initializes out)
    ffn1_kernel<<<g1, blk, 0, stream>>>(Xb, w1sB, nullptr, shared_b1, Hbuf);
    ffn2_kernel<<<g2, blk, 0, stream>>>(Hbuf, w2sB, nullptr, shared_b2, nullptr, out, 1);

    // 4) spa specialist branch (per-sample expert, gated accumulate)
    ffn1_kernel<<<g1, blk, 0, stream>>>(Xb, spaW1B, spaIdx, spa_b1, Hbuf);
    ffn2_kernel<<<g2, blk, 0, stream>>>(Hbuf, spaW2B, spaIdx, spa_b2, spaGate, out, 0);

    // 5) spe specialist branch
    ffn1_kernel<<<g1, blk, 0, stream>>>(Xb, speW1B, speIdx, spe_b1, Hbuf);
    ffn2_kernel<<<g2, blk, 0, stream>>>(Hbuf, speW2B, speIdx, spe_b2, speGate, out, 0);
}
